// PNAModel_48515950576411
// MI455X (gfx1250) — matrix operand order repaired and checked by reading, not compiled, not run on hardware
//
#include <hip/hip_runtime.h>
#include <hip/hip_bf16.h>

// ---------------------------------------------------------------------------
// PNA GNN forward for MI455X (gfx1250).  All large GEMMs run through
// v_wmma_f32_16x16x32_f16 (wave32 WMMA).  Pre-NN GEMM fuses PNA segment
// aggregation (sum / sum^2 / min / max) into its epilogue with atomics.
// ---------------------------------------------------------------------------

typedef _Float16 half8  __attribute__((ext_vector_type(8)));
typedef _Float16 half16 __attribute__((ext_vector_type(16)));
typedef float    floatx8 __attribute__((ext_vector_type(8)));

#define NN    16384
#define EE    65536
#define GG    512
#define TT    4
#define FIN   128
#define FOUT  32
#define KPRE  384      // 3*FIN
#define KPOST 1664     // 13*FIN
#define AVG_LOG 1.02273087f

#define LDSB_STRIDE 40   // halves per transposed-B column (80B, keeps b128 align)

// ---- monotonic uint encoding for float min/max atomics ---------------------
__device__ __forceinline__ unsigned fkey(float x) {
  unsigned u = __float_as_uint(x);
  return (u & 0x80000000u) ? ~u : (u | 0x80000000u);
}
__device__ __forceinline__ float fromKey(unsigned k) {
  unsigned u = (k & 0x80000000u) ? (k & 0x7fffffffu) : ~k;
  return __uint_as_float(u);
}

// ---------------------------------------------------------------------------
// WMMA GEMM core: block = 128 threads (4 waves), block tile = 64(M) x 64(N).
// Wave w computes rows [mBase+16w, +16), cols [0,64) as 4 accumulators.
// A: f16 row-major, fragments loaded directly (two b128 per lane per k-step).
// B: f16 row-major, staged *transposed* into LDS, fragments read as two
//    contiguous b128 ds reads per lane (layout per ISA 7.12.2).
// ---------------------------------------------------------------------------
__device__ __forceinline__ void gemm_tile(
    const _Float16* __restrict__ A, int lda,
    const _Float16* __restrict__ B, int ldb,
    int ndEff, int K, int mBase, _Float16* ldsBt, floatx8* acc)
{
  const int tid  = threadIdx.x;
  const int wave = tid >> 5;
  const int lane = tid & 31;
  const int n    = lane & 15;
  const int hlf  = lane >> 4;
  const int aRow = mBase + wave * 16 + n;
  const _Float16* __restrict__ arow = A + (size_t)aRow * lda;

  for (int i = 0; i < 4; ++i)
    for (int j = 0; j < 8; ++j) acc[i][j] = 0.0f;

  for (int k0 = 0; k0 < K; k0 += 32) {
    __syncthreads();
    // ---- stage B[k0..k0+31, 0..63] transposed into LDS -------------------
    #pragma unroll
    for (int c = tid; c < 256; c += 128) {
      int kr = c >> 3;            // 0..31
      int c0 = (c & 7) << 3;      // 0,8,...,56
      half8 v;
      if (c0 < ndEff) {
        v = *(const half8*)(B + (size_t)(k0 + kr) * ldb + c0);
      } else {
        for (int j = 0; j < 8; ++j) v[j] = (_Float16)0.0f;
      }
      #pragma unroll
      for (int j = 0; j < 8; ++j) ldsBt[(c0 + j) * LDSB_STRIDE + kr] = v[j];
    }
    __syncthreads();

    // ---- A fragment (ISA layout: half 0 -> K {0..7,16..23}) --------------
    union { half16 v; half8 h[2]; } af;
    af.h[0] = *(const half8*)(arow + k0 + 8 * hlf);
    af.h[1] = *(const half8*)(arow + k0 + 16 + 8 * hlf);
    if (k0 + 32 < K) __builtin_prefetch((const void*)(arow + k0 + 32), 0, 1);

    #pragma unroll
    for (int cn = 0; cn < 4; ++cn) {
      union { half16 v; half8 h[2]; } bf;
      const _Float16* bp = ldsBt + (cn * 16 + n) * LDSB_STRIDE + 8 * hlf;
      bf.h[0] = *(const half8*)(bp);
      bf.h[1] = *(const half8*)(bp + 16);
      acc[cn] = __builtin_amdgcn_wmma_f32_16x16x32_f16(
          false, af.v, false, bf.v, (short)0, acc[cn], false, false);
    }
  }
}

// ---- generic bias(+relu) GEMM, optional f32 and/or f16 outputs ------------
__global__ void __launch_bounds__(128)
gemm_bias_kernel(const _Float16* __restrict__ A, int lda,
                 const _Float16* __restrict__ B, int ldb,
                 const float* __restrict__ bias,
                 float* Cf, _Float16* Ch, int ldc, int colOff,
                 int M, int Nd, int K, int doRelu)
{
  __shared__ _Float16 ldsBt[64 * LDSB_STRIDE];
  floatx8 acc[4];
  const int mBase = blockIdx.y * 64;
  const int nBase = blockIdx.x * 64;
  gemm_tile(A, lda, B + nBase, ldb, Nd - nBase, K, mBase, ldsBt, acc);

  const int lane = threadIdx.x & 31, wave = threadIdx.x >> 5;
  const int n = lane & 15, hlf = lane >> 4;
  #pragma unroll
  for (int cn = 0; cn < 4; ++cn) {
    int col = nBase + cn * 16 + n;
    if (col >= Nd) continue;
    float bv = bias ? bias[col] : 0.0f;
    int gcol = colOff + col;
    #pragma unroll
    for (int v = 0; v < 8; ++v) {
      int row = mBase + wave * 16 + v + 8 * hlf;
      float val = acc[cn][v] + bv;
      if (doRelu) val = fmaxf(val, 0.0f);
      if (Cf) Cf[(size_t)row * ldc + gcol] = val;
      if (Ch) Ch[(size_t)row * ldc + gcol] = (_Float16)val;
    }
  }
}

// ---- pre-NN GEMM with fused PNA segment aggregation epilogue --------------
__global__ void __launch_bounds__(128)
gemm_pre_agg_kernel(const _Float16* __restrict__ A /*hcat [E,384]*/,
                    const _Float16* __restrict__ B /*Wpre[l][t] [384,128]*/,
                    const float* __restrict__ bias /*bpre [128]*/,
                    const int* __restrict__ dstIdx,
                    float* sumB, float* sqB, unsigned* minB, unsigned* maxB,
                    int tower)
{
  __shared__ _Float16 ldsBt[64 * LDSB_STRIDE];
  floatx8 acc[4];
  const int mBase = blockIdx.y * 64;
  const int nBase = blockIdx.x * 64;
  gemm_tile(A, KPRE, B + nBase, FIN, FIN - nBase, KPRE, mBase, ldsBt, acc);

  const int lane = threadIdx.x & 31, wave = threadIdx.x >> 5;
  const int n = lane & 15, hlf = lane >> 4;
  #pragma unroll
  for (int cn = 0; cn < 4; ++cn) {
    int col = nBase + cn * 16 + n;          // 0..127
    float bv = bias[col];
    #pragma unroll
    for (int v = 0; v < 8; ++v) {
      int row = mBase + wave * 16 + v + 8 * hlf;   // edge id
      int d   = dstIdx[row];
      float val = acc[cn][v] + bv;
      size_t off = ((size_t)d * TT + tower) * FIN + col;
      atomicAdd(sumB + off, val);
      atomicAdd(sqB  + off, val * val);
      unsigned key = fkey(val);
      atomicMin(minB + off, key);
      atomicMax(maxB + off, key);
    }
  }
}

// ---------------------------------------------------------------------------
// Scalar kernels
// ---------------------------------------------------------------------------
__global__ void cvt_f16_kernel(const float* in, _Float16* out, int nElem) {
  int i = blockIdx.x * blockDim.x + threadIdx.x;
  if (i < nElem) out[i] = (_Float16)in[i];
}
__global__ void fill_f32_kernel(float* p, float v, int nElem) {
  int i = blockIdx.x * blockDim.x + threadIdx.x;
  if (i < nElem) p[i] = v;
}
__global__ void fill_u32_kernel(unsigned* p, unsigned v, int nElem) {
  int i = blockIdx.x * blockDim.x + threadIdx.x;
  if (i < nElem) p[i] = v;
}

__global__ void node_enc_kernel(const float* __restrict__ x,
                                const float* __restrict__ Wa,
                                const float* __restrict__ ba,
                                float* h, _Float16* hh) {
  int i = blockIdx.x * blockDim.x + threadIdx.x;
  if (i >= NN * FIN) return;
  int nI = i >> 7, f = i & 127;
  float acc = ba[f];
  #pragma unroll
  for (int k = 0; k < 11; ++k) acc += x[nI * 11 + k] * Wa[k * FIN + f];
  acc = fmaxf(acc, 0.0f);
  h[i] = acc; hh[i] = (_Float16)acc;
}

__global__ void edge_enc_kernel(const float* __restrict__ ea,
                                const float* __restrict__ We,
                                const float* __restrict__ be,
                                _Float16* out) {
  int i = blockIdx.x * blockDim.x + threadIdx.x;
  if (i >= EE * FIN) return;
  int e = i >> 7, f = i & 127;
  float acc = be[f];
  #pragma unroll
  for (int k = 0; k < 4; ++k) acc += ea[e * 4 + k] * We[k * FIN + f];
  out[i] = (_Float16)fmaxf(acc, 0.0f);
}

__global__ void deg_kernel(const int* __restrict__ ei, float* deg) {
  int e = blockIdx.x * blockDim.x + threadIdx.x;
  if (e < EE) atomicAdd(&deg[ei[EE + e]], 1.0f);
}

__global__ void hcat_kernel(const int* __restrict__ ei,
                            const _Float16* __restrict__ hh,
                            const _Float16* __restrict__ eal,
                            _Float16* __restrict__ hcat) {
  int i = blockIdx.x * blockDim.x + threadIdx.x;
  if (i >= EE * FIN) return;
  int e = i >> 7, f = i & 127;
  int s = ei[e], d = ei[EE + e];
  _Float16* row = hcat + (size_t)e * KPRE;
  row[f]       = hh[(size_t)d * FIN + f];
  row[128 + f] = hh[(size_t)s * FIN + f];
  row[256 + f] = eal[i];
}

// build post-NN input row [xt | agg | agg*s | agg/s] for one tower
__global__ void finalize_kernel(const float* __restrict__ sumB,
                                const float* __restrict__ sqB,
                                const unsigned* __restrict__ minB,
                                const unsigned* __restrict__ maxB,
                                const float* __restrict__ deg,
                                const float* __restrict__ h,
                                int tower, _Float16* __restrict__ postin) {
  int i = blockIdx.x * blockDim.x + threadIdx.x;
  if (i >= NN * FIN) return;
  int nI = i >> 7, f = i & 127;
  size_t off = ((size_t)nI * TT + tower) * FIN + f;
  float dc   = fmaxf(deg[nI], 1.0f);
  float inv  = 1.0f / dc;
  float mean = sumB[off] * inv;
  float m2   = sqB[off] * inv;
  float stdv = sqrtf(fmaxf(m2 - mean * mean, 0.0f) + 1e-5f);
  unsigned mk = minB[off], xk = maxB[off];
  float mn = (mk == 0xFFFFFFFFu) ? 0.0f : fromKey(mk);
  float mx = (xk == 0u) ? 0.0f : fromKey(xk);
  float s  = logf(dc + 1.0f) / AVG_LOG;
  float is = 1.0f / s;
  _Float16* row = postin + (size_t)nI * KPOST;
  row[f]          = (_Float16)h[i];
  row[128  + f] = (_Float16)mean; row[256  + f] = (_Float16)mn;
  row[384  + f] = (_Float16)mx;   row[512  + f] = (_Float16)stdv;
  row[640  + f] = (_Float16)(mean * s);  row[768  + f] = (_Float16)(mn * s);
  row[896  + f] = (_Float16)(mx * s);    row[1024 + f] = (_Float16)(stdv * s);
  row[1152 + f] = (_Float16)(mean * is); row[1280 + f] = (_Float16)(mn * is);
  row[1408 + f] = (_Float16)(mx * is);   row[1536 + f] = (_Float16)(stdv * is);
}

// BatchNorm (training-mode biased stats): one block per feature
__global__ void __launch_bounds__(256)
bn_reduce_kernel(const float* __restrict__ X, int rows, int feats,
                 float* meanv, float* varv) {
  __shared__ float s1[256], s2[256];
  int f = blockIdx.x, t = threadIdx.x;
  float a = 0.0f, b = 0.0f;
  for (int r = t; r < rows; r += 256) {
    float v = X[(size_t)r * feats + f];
    a += v; b += v * v;
  }
  s1[t] = a; s2[t] = b; __syncthreads();
  for (int s = 128; s > 0; s >>= 1) {
    if (t < s) { s1[t] += s1[t + s]; s2[t] += s2[t + s]; }
    __syncthreads();
  }
  if (t == 0) {
    float m = s1[0] / rows;
    meanv[f] = m;
    varv[f]  = s2[0] / rows - m * m;
  }
}
__global__ void bn_apply_kernel(const float* __restrict__ X,
                                const float* __restrict__ meanv,
                                const float* __restrict__ varv,
                                const float* __restrict__ g,
                                const float* __restrict__ b,
                                int rows, int feats,
                                float* outF, _Float16* outH, int doRelu) {
  int i = blockIdx.x * blockDim.x + threadIdx.x;
  if (i >= rows * feats) return;
  int f = i % feats;
  float v = (X[i] - meanv[f]) * rsqrtf(varv[f] + 1e-5f) * g[f] + b[f];
  if (doRelu) v = fmaxf(v, 0.0f);
  if (outF) outF[i] = v;
  if (outH) outH[i] = (_Float16)v;
}

__global__ void pool_cnt_kernel(const int* __restrict__ batch, float* cnt) {
  int n = blockIdx.x * blockDim.x + threadIdx.x;
  if (n < NN) atomicAdd(&cnt[batch[n]], 1.0f);
}
__global__ void pool_sum_kernel(const int* __restrict__ batch,
                                const float* __restrict__ h, float* pooled) {
  int i = blockIdx.x * blockDim.x + threadIdx.x;
  if (i >= NN * FIN) return;
  int n = i >> 7, f = i & 127;
  atomicAdd(&pooled[(size_t)batch[n] * FIN + f], h[i]);
}
__global__ void head1_kernel(const float* __restrict__ pooled,
                             const float* __restrict__ cnt,
                             const float* __restrict__ W1,
                             const float* __restrict__ b1, float* z) {
  int i = blockIdx.x * blockDim.x + threadIdx.x;
  if (i >= GG * 64) return;
  int g = i >> 6, j = i & 63;
  float c = fmaxf(cnt[g], 1.0f);
  float acc = b1[j];
  #pragma unroll 8
  for (int k = 0; k < FIN; ++k) acc += (pooled[g * FIN + k] / c) * W1[k * 64 + j];
  z[i] = fmaxf(acc, 0.0f);
}
__global__ void head2_kernel(const float* __restrict__ zbn,
                             const float* __restrict__ W2,
                             const float* __restrict__ b2, float* out) {
  int g = blockIdx.x * blockDim.x + threadIdx.x;
  if (g >= GG) return;
  float acc = b2[0];
  #pragma unroll 8
  for (int j = 0; j < 64; ++j) acc += zbn[g * 64 + j] * W2[j];
  out[g] = acc;
}

// ---------------------------------------------------------------------------
extern "C" void kernel_launch(void* const* d_in, const int* in_sizes, int n_in,
                              void* d_out, int out_size, void* d_ws, size_t ws_size,
                              hipStream_t stream) {
  const float* x        = (const float*)d_in[0];
  const float* ea_raw   = (const float*)d_in[1];
  const int*   ei       = (const int*)d_in[2];
  const int*   batch    = (const int*)d_in[3];
  const float* Wa       = (const float*)d_in[4];
  const float* ba       = (const float*)d_in[5];
  const float* We       = (const float*)d_in[6];
  const float* be       = (const float*)d_in[7];
  const float* We_conv  = (const float*)d_in[8];
  const float* be_conv  = (const float*)d_in[9];
  const float* Wpre     = (const float*)d_in[10];
  const float* bpre     = (const float*)d_in[11];
  const float* Wpost    = (const float*)d_in[12];
  const float* bpost    = (const float*)d_in[13];
  const float* Wlin     = (const float*)d_in[14];
  const float* blin     = (const float*)d_in[15];
  const float* bng      = (const float*)d_in[16];
  const float* bnb      = (const float*)d_in[17];
  const float* W1       = (const float*)d_in[18];
  const float* b1       = (const float*)d_in[19];
  const float* hg       = (const float*)d_in[20];
  const float* hb       = (const float*)d_in[21];
  const float* W2       = (const float*)d_in[22];
  const float* b2       = (const float*)d_in[23];
  float* out = (float*)d_out;

  // ---- workspace carve ---------------------------------------------------
  char* base = (char*)d_ws;
  size_t off = 0;
  auto carve = [&](size_t bytes) -> char* {
    char* r = base + off;
    off += (bytes + 255) & ~(size_t)255;
    return r;
  };
  float*     h_f32  = (float*)    carve((size_t)NN * FIN * 4);
  _Float16*  h_f16  = (_Float16*) carve((size_t)NN * FIN * 2);
  float*     tmp    = (float*)    carve((size_t)NN * FIN * 4);
  _Float16*  hpost  = (_Float16*) carve((size_t)NN * FIN * 2);
  _Float16*  ea_h   = (_Float16*) carve((size_t)EE * FIN * 2);
  _Float16*  ea_l   = (_Float16*) carve((size_t)EE * FIN * 2);
  _Float16*  hcat   = (_Float16*) carve((size_t)EE * KPRE * 2);
  float*     sumB   = (float*)    carve((size_t)NN * TT * FIN * 4);
  float*     sqB    = (float*)    carve((size_t)NN * TT * FIN * 4);
  unsigned*  minB   = (unsigned*) carve((size_t)NN * TT * FIN * 4);
  unsigned*  maxB   = (unsigned*) carve((size_t)NN * TT * FIN * 4);
  float*     degB   = (float*)    carve((size_t)NN * 4);
  _Float16*  postin = (_Float16*) carve((size_t)NN * KPOST * 2);
  _Float16*  WconvH = (_Float16*) carve((size_t)3 * FIN * FIN * 2);
  _Float16*  WpreH  = (_Float16*) carve((size_t)12 * KPRE * FIN * 2);
  _Float16*  WpostH = (_Float16*) carve((size_t)12 * KPOST * FOUT * 2);
  _Float16*  WlinH  = (_Float16*) carve((size_t)3 * FIN * FIN * 2);
  float*     meanv  = (float*)    carve(FIN * 4);
  float*     varv   = (float*)    carve(FIN * 4);
  float*     pooled = (float*)    carve((size_t)GG * FIN * 4);
  float*     cnt    = (float*)    carve(GG * 4);
  float*     z      = (float*)    carve((size_t)GG * 64 * 4);
  float*     zbn    = (float*)    carve((size_t)GG * 64 * 4);
  (void)ws_size; (void)in_sizes; (void)n_in; (void)out_size;

  auto blocks = [](long n, int b) { return dim3((unsigned)((n + b - 1) / b)); };

  // ---- weight conversion to f16 -----------------------------------------
  cvt_f16_kernel<<<blocks(3 * FIN * FIN, 256), 256, 0, stream>>>(We_conv, WconvH, 3 * FIN * FIN);
  cvt_f16_kernel<<<blocks(12 * KPRE * FIN, 256), 256, 0, stream>>>(Wpre, WpreH, 12 * KPRE * FIN);
  cvt_f16_kernel<<<blocks(12 * KPOST * FOUT, 256), 256, 0, stream>>>(Wpost, WpostH, 12 * KPOST * FOUT);
  cvt_f16_kernel<<<blocks(3 * FIN * FIN, 256), 256, 0, stream>>>(Wlin, WlinH, 3 * FIN * FIN);

  // ---- encoders + degree -------------------------------------------------
  node_enc_kernel<<<blocks((long)NN * FIN, 256), 256, 0, stream>>>(x, Wa, ba, h_f32, h_f16);
  edge_enc_kernel<<<blocks((long)EE * FIN, 256), 256, 0, stream>>>(ea_raw, We, be, ea_h);
  fill_f32_kernel<<<blocks(NN, 256), 256, 0, stream>>>(degB, 0.0f, NN);
  deg_kernel<<<blocks(EE, 256), 256, 0, stream>>>(ei, degB);

  const long aggElems = (long)NN * TT * FIN;

  for (int l = 0; l < 3; ++l) {
    // conv-internal edge encoder: ea_l = ea_h @ We_conv[l] + be_conv[l]
    gemm_bias_kernel<<<dim3(2, EE / 64), 128, 0, stream>>>(
        ea_h, FIN, WconvH + (size_t)l * FIN * FIN, FIN,
        be_conv + (size_t)l * FIN, nullptr, ea_l, FIN, 0,
        EE, FIN, FIN, 0);

    // hcat = [h[dst] | h[src] | ea_l]
    hcat_kernel<<<blocks((long)EE * FIN, 256), 256, 0, stream>>>(ei, h_f16, ea_l, hcat);

    // reset aggregation buffers
    fill_f32_kernel<<<blocks(aggElems, 256), 256, 0, stream>>>(sumB, 0.0f, (int)aggElems);
    fill_f32_kernel<<<blocks(aggElems, 256), 256, 0, stream>>>(sqB, 0.0f, (int)aggElems);
    fill_u32_kernel<<<blocks(aggElems, 256), 256, 0, stream>>>(minB, 0xFFFFFFFFu, (int)aggElems);
    fill_u32_kernel<<<blocks(aggElems, 256), 256, 0, stream>>>(maxB, 0u, (int)aggElems);

    // pre-NN GEMM per tower, fused segment aggregation epilogue
    for (int t = 0; t < TT; ++t) {
      gemm_pre_agg_kernel<<<dim3(2, EE / 64), 128, 0, stream>>>(
          hcat, WpreH + ((size_t)l * TT + t) * KPRE * FIN,
          bpre + ((size_t)l * TT + t) * FIN, ei + EE,
          sumB, sqB, minB, maxB, t);
    }

    // finalize aggregates -> postin, then post-NN GEMM into hpost[:, t*32..]
    for (int t = 0; t < TT; ++t) {
      finalize_kernel<<<blocks((long)NN * FIN, 256), 256, 0, stream>>>(
          sumB, sqB, minB, maxB, degB, h_f32, t, postin);
      gemm_bias_kernel<<<dim3(1, NN / 64), 128, 0, stream>>>(
          postin, KPOST, WpostH + ((size_t)l * TT + t) * KPOST * FOUT, FOUT,
          bpost + ((size_t)l * TT + t) * FOUT, nullptr, hpost, FIN, t * FOUT,
          NN, FOUT, KPOST, 0);
    }

    // tower mixing: tmp = hpost @ Wlin[l] + blin[l]
    gemm_bias_kernel<<<dim3(2, NN / 64), 128, 0, stream>>>(
        hpost, FIN, WlinH + (size_t)l * FIN * FIN, FIN,
        blin + (size_t)l * FIN, tmp, nullptr, FIN, 0,
        NN, FIN, FIN, 0);

    // BatchNorm + ReLU -> next layer h
    bn_reduce_kernel<<<FIN, 256, 0, stream>>>(tmp, NN, FIN, meanv, varv);
    bn_apply_kernel<<<blocks((long)NN * FIN, 256), 256, 0, stream>>>(
        tmp, meanv, varv, bng + (size_t)l * FIN, bnb + (size_t)l * FIN,
        NN, FIN, h_f32, h_f16, 1);
  }

  // ---- global mean pool + head MLP --------------------------------------
  fill_f32_kernel<<<blocks((long)GG * FIN, 256), 256, 0, stream>>>(pooled, 0.0f, GG * FIN);
  fill_f32_kernel<<<blocks(GG, 256), 256, 0, stream>>>(cnt, 0.0f, GG);
  pool_cnt_kernel<<<blocks(NN, 256), 256, 0, stream>>>(batch, cnt);
  pool_sum_kernel<<<blocks((long)NN * FIN, 256), 256, 0, stream>>>(batch, h_f32, pooled);

  head1_kernel<<<blocks((long)GG * 64, 256), 256, 0, stream>>>(pooled, cnt, W1, b1, z);
  bn_reduce_kernel<<<64, 256, 0, stream>>>(z, GG, 64, meanv, varv);
  bn_apply_kernel<<<blocks((long)GG * 64, 256), 256, 0, stream>>>(
      z, meanv, varv, hg, hb, GG, 64, zbn, nullptr, 0);
  head2_kernel<<<blocks(GG, 256), 256, 0, stream>>>(zbn, W2, b2, out);
}
